// HDCNN_66108136620366
// MI455X (gfx1250) — matrix operand order repaired and checked by reading, not compiled, not run on hardware
//
#include <hip/hip_runtime.h>
#include <math.h>

typedef __attribute__((ext_vector_type(4)))  _Float16 v4h;
typedef __attribute__((ext_vector_type(8)))  _Float16 v8h;
typedef __attribute__((ext_vector_type(16))) _Float16 v16h;
typedef __attribute__((ext_vector_type(4)))  float    v4f;
typedef __attribute__((ext_vector_type(8)))  float    v8f;

#define BATCH       4096
#define INLEN       8192
#define FLT         512
#define OUTSTRIDE   10240     // max row length (layer 3), used for all intermediate buffers
#define SPAN        512       // output columns produced per workgroup
#define AW          1056      // SPAN + 512 (left halo) + 32 (chunk tail); multiple of 32
#define NCHUNK      17        // ceil(544/32) K-chunks of 32

// ---------------------------------------------------------------------------
// Conv layer as WMMA GEMM with Toeplitz filter matrix.
// conv[b, j] = sum_{k=0..511} v[b, j-k] * w[k], v zero-extended outside [0,8192).
// Tile: M=16 batch rows, N=16 output cols, K = t-index in [-512,31] (17 x 32).
//   A[m,k] = v[b0+m, j0 + t0 + k]          (f16, from LDS sliding window)
//   B[k,n] = w[n - t0 - k]                 (f16 Toeplitz; all 17 chunks hoisted
//                                           to VGPRs once per wave)
// Also accumulates per-row  sq = sum(conv^2)  and  dp = sum(conv*bk).
// ---------------------------------------------------------------------------
__global__ __launch_bounds__(256)
void hdcnn_conv_wmma(const float* __restrict__ in, int inStride,
                     const float* __restrict__ w,      // 512 taps of this layer
                     const float* __restrict__ bk,     // bias row, length Lout
                     float* __restrict__ convOut,      // stride OUTSTRIDE
                     float* __restrict__ sqg,          // [BATCH] sum conv^2
                     float* __restrict__ dpg,          // [BATCH] sum conv*bk
                     int Lout)
{
    __shared__ __align__(16) _Float16 As[16 * AW];          // 33,792 B
    __shared__ __align__(16) _Float16 Bs[NCHUNK * 16 * 32]; // 17,408 B (BsT[c][n][k])
    __shared__ float sqs[16], dps[16];

    const int tid  = threadIdx.x;
    const int lane = tid & 31;
    const int wave = tid >> 5;                 // 8 waves
    const int b0   = blockIdx.x * 16;
    const int j0base  = blockIdx.y * SPAN;
    const int colBase = j0base - 512;          // leftmost input column cached

    if (tid < 16) { sqs[tid] = 0.0f; dps[tid] = 0.0f; }

    // ---- stage A: 16 rows x AW cols, f32 -> f16, vectorized x4 ----
    // colBase is a multiple of 512 and each thread handles 4 consecutive
    // columns, so the fast path is one global b128 load + one ds_store_b64.
    for (int m = 0; m < 16; ++m) {
        const float* src = in + (size_t)(b0 + m) * inStride;
        for (int cc = tid * 4; cc < AW; cc += 1024) {
            const int col = colBase + cc;
            v4f v;
            if (col >= 0 && col + 3 < INLEN) {
                v = *(const v4f*)(src + col);
            } else {
                #pragma unroll
                for (int e = 0; e < 4; ++e)
                    v[e] = (col + e >= 0 && col + e < INLEN) ? src[col + e] : 0.0f;
            }
            v4h h;
            #pragma unroll
            for (int e = 0; e < 4; ++e) h[e] = (_Float16)v[e];
            *(v4h*)&As[m * AW + cc] = h;
        }
    }
    // ---- stage Toeplitz B (transposed: BsT[c][n][k] = w[n + 512 - 32c - k]) ----
    for (int idx = tid; idx < NCHUNK * 512; idx += 256) {
        int c = idx >> 9, rem = idx & 511;
        int n = rem >> 5, k = rem & 31;
        int wi = n + 512 - 32 * c - k;
        Bs[idx] = (_Float16)((wi >= 0 && wi < FLT) ? w[wi] : 0.0f);
    }
    __syncthreads();

    const int g  = lane >> 4;      // lane group (K-half selector)
    const int mr = lane & 15;      // A row M / B col N

    // ---- hoist all 17 B fragments into VGPRs (shared by this wave's tiles) ----
    // B fragment layout: lane holds N=mr, K in {8g..8g+7} U {16+8g..16+8g+7}.
    v16h bfr[NCHUNK];
    #pragma unroll
    for (int c = 0; c < NCHUNK; ++c) {
        const _Float16* bp = &Bs[(c * 16 + mr) * 32 + 8 * g];
        v8h blo = *(const v8h*)bp;
        v8h bhi = *(const v8h*)(bp + 16);
        bfr[c] = __builtin_shufflevector(blo, bhi,
                 0,1,2,3,4,5,6,7, 8,9,10,11,12,13,14,15);
    }

    float sqa[8] = {0,0,0,0,0,0,0,0};
    float dpa[8] = {0,0,0,0,0,0,0,0};

    for (int tile = wave; tile < SPAN / 16; tile += 8) {
        const int j0 = j0base + tile * 16;
        if (j0 >= Lout) break;                 // wave-uniform: EXEC stays full for WMMA

        v8f acc = {};
        // A fragment: lane holds M=mr, K in {8g..8g+7} U {16+8g..16+8g+7}
        const int acol0 = (j0 - colBase) - 512 + 8 * g;   // chunk 0, run 0
        #pragma unroll
        for (int c = 0; c < NCHUNK; ++c) {
            const _Float16* ap = &As[mr * AW + acol0 + 32 * c];
            v8h alo = *(const v8h*)ap;          // K = 8g .. 8g+7      (b128, 16B aligned)
            v8h ahi = *(const v8h*)(ap + 16);   // K = 16+8g .. 16+8g+7
            v16h a = __builtin_shufflevector(alo, ahi,
                     0,1,2,3,4,5,6,7, 8,9,10,11,12,13,14,15);
            acc = __builtin_amdgcn_wmma_f32_16x16x32_f16(
                      false, a, false, bfr[c], (short)0, acc, false, false);
        }
        // D layout: VGPR r -> row M = r + 8g, col N = mr
        #pragma unroll
        for (int r = 0; r < 8; ++r) {
            int j = j0 + mr;
            if (j < Lout) {
                float d = acc[r];
                convOut[(size_t)(b0 + r + 8 * g) * OUTSTRIDE + j] = d;
                sqa[r] += d * d;
                dpa[r] += d * bk[j];
            }
        }
    }
    // per-row reductions: registers -> LDS atomics -> global atomics
    #pragma unroll
    for (int r = 0; r < 8; ++r) {
        atomicAdd(&sqs[r + 8 * g], sqa[r]);
        atomicAdd(&dps[r + 8 * g], dpa[r]);
    }
    __syncthreads();
    if (tid < 16) {
        atomicAdd(&sqg[b0 + tid], sqs[tid]);
        atomicAdd(&dpg[b0 + tid], dps[tid]);
    }
}

// ---------------------------------------------------------------------------
// y2 = sum(bk^2) over the full (padded) row — one small workgroup.
// ---------------------------------------------------------------------------
__global__ __launch_bounds__(256)
void hdcnn_y2(const float* __restrict__ bk, int L, float* __restrict__ y2buf)
{
    __shared__ float red[256];
    float s = 0.0f;
    for (int j = threadIdx.x; j < L; j += 256) { float v = bk[j]; s += v * v; }
    red[threadIdx.x] = s; __syncthreads();
    for (int off = 128; off > 0; off >>= 1) {
        if ((int)threadIdx.x < off) red[threadIdx.x] += red[threadIdx.x + off];
        __syncthreads();
    }
    if (threadIdx.x == 0) y2buf[0] = red[0];
}

// ---------------------------------------------------------------------------
// Per-row scalars: expmap0 scale s, proj clip, mobius_add coefficients.
// out[b,j] = relu( p[b]*conv[b,j] + q[b]*bk[j] )  with
//   p = alpha*s/gamma, q = beta/gamma.
// ---------------------------------------------------------------------------
__global__ __launch_bounds__(256)
void hdcnn_scalars(const float* __restrict__ sqg, const float* __restrict__ dpg,
                   const float* __restrict__ y2buf, float* __restrict__ pq)
{
    int b = blockIdx.x * 256 + threadIdx.x;
    if (b >= BATCH) return;
    const float c  = 1e-12f;
    const float sc = 1e-6f;                 // sqrt(c)
    float n2 = sqg[b];
    float un = fmaxf(sqrtf(n2), 1e-15f);    // ||conv||
    float a1 = sc * un;
    float s1 = tanhf(a1) / a1;              // expmap0 scale
    float hn = s1 * un;                     // ||hyp|| pre-proj
    const float maxn = (1.0f - 0.004f) / sc;
    float s2 = (hn > maxn) ? (maxn / hn) : 1.0f;
    float s  = s1 * s2;
    float xn = s * un;                      // final ||x||
    float x2 = xn * xn;
    float xy = s * dpg[b];
    float y2 = y2buf[0];
    float al = 1.0f + 2.0f * c * xy + c * y2;
    float be = 1.0f - c * x2;
    float dn = 1.0f + 2.0f * c * xy + c * c * x2 * y2;
    float ga = fmaxf(dn, 1e-15f);
    pq[2 * b]     = al * s / ga;
    pq[2 * b + 1] = be / ga;
}

// ---------------------------------------------------------------------------
// Elementwise mobius_add + relu (safe in-place: each thread RMWs its element).
// ---------------------------------------------------------------------------
__global__ __launch_bounds__(256)
void hdcnn_pointwise(const float* __restrict__ conv, const float* __restrict__ bk,
                     const float* __restrict__ pq, float* __restrict__ out, int Lout)
{
    int b = blockIdx.y;
    int j = blockIdx.x * 256 + threadIdx.x;
    if (j >= Lout) return;
    float v = pq[2 * b] * conv[(size_t)b * OUTSTRIDE + j] + pq[2 * b + 1] * bk[j];
    out[(size_t)b * OUTSTRIDE + j] = fmaxf(v, 0.0f);
}

// ---------------------------------------------------------------------------
extern "C" void kernel_launch(void* const* d_in, const int* in_sizes, int n_in,
                              void* d_out, int out_size, void* d_ws, size_t ws_size,
                              hipStream_t stream)
{
    (void)in_sizes; (void)n_in; (void)out_size; (void)ws_size;
    const float* hk = (const float*)d_in[0];
    const float* w  = (const float*)d_in[1];
    const float* bks[4] = { (const float*)d_in[2], (const float*)d_in[3],
                            (const float*)d_in[4], (const float*)d_in[5] };
    float* out  = (float*)d_out;

    float* bufB = (float*)d_ws;                           // 4096 x 10240 f32 (160 MB)
    float* sqg  = bufB + (size_t)BATCH * OUTSTRIDE;       // [4096]
    float* dpg  = sqg + BATCH;                            // [4096]
    float* pq   = dpg + BATCH;                            // [2*4096]
    float* y2b  = pq + 2 * BATCH;                         // [1]

    // Ping-pong: layer conv target alternates bufB <-> d_out; pointwise is
    // in-place, so layer i+1 reads the just-finished buffer (L2-resident).
    const float* src[4] = { hk,   bufB, out,  bufB };
    float*       dst[4] = { bufB, out,  bufB, out  };
    const int strideIn[4] = { INLEN, OUTSTRIDE, OUTSTRIDE, OUTSTRIDE };

    for (int i = 0; i < 4; ++i) {
        const int Li = INLEN + (i + 1) * FLT;             // 8704, 9216, 9728, 10240
        hipMemsetAsync(sqg, 0, 2 * BATCH * sizeof(float), stream);
        hipLaunchKernelGGL(hdcnn_y2, dim3(1), dim3(256), 0, stream, bks[i], Li, y2b);
        dim3 gconv(BATCH / 16, (Li + SPAN - 1) / SPAN);
        hipLaunchKernelGGL(hdcnn_conv_wmma, gconv, dim3(256), 0, stream,
                           src[i], strideIn[i], w + i * FLT, bks[i],
                           dst[i], sqg, dpg, Li);
        hipLaunchKernelGGL(hdcnn_scalars, dim3(BATCH / 256), dim3(256), 0, stream,
                           sqg, dpg, y2b, pq);
        dim3 gpw((Li + 255) / 256, BATCH);
        hipLaunchKernelGGL(hdcnn_pointwise, gpw, dim3(256), 0, stream,
                           dst[i], bks[i], pq, dst[i], Li);
    }
}
